// NormPermutePool_59133109732020
// MI455X (gfx1250) — compile-verified, matrix operand order checked by reading, not run on hardware
//
#include <hip/hip_runtime.h>
#include <hip/hip_bf16.h>
#include <hip/hip_fp16.h>

// ---------------------------------------------------------------------------
// NormPermutePool on MI455X (gfx1250):
//   features (f16, built on device) @ W^T (f32 -> f16 once) with f32 accum
//   via v_wmma_f32_16x16x32_f16.  W stream (134 MB) is the HBM floor; A+Wh
//   (131 MB) are L2-resident.  32x64 register blocking per wave lifts L2
//   arithmetic intensity 8 -> 21.8 FLOP/byte (2.1 GB -> 0.8 GB of L2 reads).
//   Addresses are derived straight from kernel-arg bases (no pointer arrays)
//   so loads lower to GLOBAL (loadcnt-only), not FLAT (loadcnt+dscnt).
// ---------------------------------------------------------------------------

typedef __attribute__((ext_vector_type(16))) _Float16 v16h;
typedef __attribute__((ext_vector_type(8)))  _Float16 v8h;
typedef __attribute__((ext_vector_type(4)))  _Float16 v4h;
typedef __attribute__((ext_vector_type(8)))  float    v8f;
typedef __attribute__((ext_vector_type(4)))  float    v4f;

#define BDIM   64
#define CDIM   4
#define RDIM   64
#define EDIM   512
#define PDIM   4
#define OSZ    256
#define MDIM   256                 /* B*C                       */
#define KTOT   131072              /* R*E*P                     */
#define LRE    32768               /* R*E (one piece)           */
#define KCH    64                  /* K split for parallelism   */
#define KSTEP  (KTOT / KCH)        /* 2048 -> 64 wmma k-steps   */
#define MT     2                   /* 16-row subtiles per wave  */
#define NT     4                   /* 16-col subtiles per wave  */

// ---------------------------------------------------------------------------
// 1) W (f32, [OSZ][KTOT]) -> Wh (f16), 4 elements / thread, b128 load.
// ---------------------------------------------------------------------------
__global__ __launch_bounds__(256) void wcvt_kernel(const float* __restrict__ W,
                                                   _Float16* __restrict__ Wh,
                                                   int n4) {
  int i = blockIdx.x * 256 + threadIdx.x;
  if (i < n4) {
    v4f w = *(const v4f*)(W + (size_t)i * 4);
    v4h h;
    h.x = (_Float16)w.x; h.y = (_Float16)w.y;
    h.z = (_Float16)w.z; h.w = (_Float16)w.w;
    *(v4h*)(Wh + (size_t)i * 4) = h;
  }
}

// ---------------------------------------------------------------------------
// 2) Feature build: block = one (m, r) row, thread = e (512 threads).
//    v[e] = x[b,c,rxd_perm[r], e_idx[e,p]] * signs[e,p]; groups of k=2^p get
//    first-argmax sparsification (k<=8 divides 512, so groups never straddle
//    rows and the reference's -inf padding is a no-op).  f16 out, row-major.
// ---------------------------------------------------------------------------
__global__ __launch_bounds__(512) void feat_kernel(const float* __restrict__ x,
                                                   const int*   __restrict__ rxd_perm,
                                                   const int*   __restrict__ e_idx,
                                                   const float* __restrict__ signs,
                                                   _Float16*    __restrict__ A) {
  __shared__ float vbuf[EDIM];
  const int mr = blockIdx.x;
  const int m  = mr >> 6;          // b*C + c
  const int r  = mr & 63;
  const int e  = threadIdx.x;
  const int rp = rxd_perm[r];
  const float*  xrow = x + ((size_t)m * RDIM + rp) * EDIM;
  _Float16*     arow = A + (size_t)m * KTOT + (size_t)r * EDIM + e;

  #pragma unroll
  for (int p = 0; p < PDIM; ++p) {
    const float v = xrow[e_idx[e * PDIM + p]] * signs[e * PDIM + p];
    __syncthreads();               // protect vbuf from previous iteration
    vbuf[e] = v;
    __syncthreads();
    const int k = 1 << p;
    float feat = v;
    if (k > 1) {
      const int g0 = e & ~(k - 1);
      float mx = vbuf[g0];
      int arg = g0;
      for (int j = 1; j < k; ++j) {            // first-argmax (strict >)
        float w2 = vbuf[g0 + j];
        if (w2 > mx) { mx = w2; arg = g0 + j; }
      }
      feat = (e == arg) ? v : 0.0f;
    }
    arow[(size_t)p * LRE] = (_Float16)feat;
  }
}

// ---------------------------------------------------------------------------
// 3) GEMM: one wave per (n-block, m-block, k-chunk); each wave owns a 32x64
//    output block = MT x NT subtiles of 16x16 with 8 f32 accumulators.
//    Per k-step: 2 A-tiles + 4 B-tiles (12 x global_load_b128, per the 16-bit
//    WMMA A/B VGPR layout) feed 8 v_wmma_f32_16x16x32_f16 -> 1.5 loads/wmma.
//    Layout: lane L -> l=L%16 (row of A / col of B), h=L/16 selects K-halves:
//    v16h[0..7] = K(h*8 + 0..7), v16h[8..15] = K(16 + h*8 + 0..7).
//    Partials to ws (no atomics -> deterministic).
// ---------------------------------------------------------------------------
__global__ __launch_bounds__(32) void gemm_kernel(const _Float16* __restrict__ A,
                                                  const _Float16* __restrict__ Wh,
                                                  float* __restrict__ part) {
  const int nt0 = blockIdx.x * (NT * 16);
  const int mt0 = blockIdx.y * (MT * 16);
  const int ch  = blockIdx.z;
  const int L   = threadIdx.x;
  const int l   = L & 15;
  const int h   = L >> 4;

  // Per-lane element offsets from the A / Wh bases (no pointer arrays, so the
  // address-space of every load stays traceable to the global kernel args).
  const size_t koff = (size_t)ch * KSTEP + h * 8;
  size_t aoff[MT];
  size_t boff[NT];
  #pragma unroll
  for (int mi = 0; mi < MT; ++mi)
    aoff[mi] = (size_t)(mt0 + mi * 16 + l) * KTOT + koff;
  #pragma unroll
  for (int ni = 0; ni < NT; ++ni)
    boff[ni] = (size_t)(nt0 + ni * 16 + l) * KTOT + koff;

  v8f acc[MT][NT] = {};

  #pragma unroll 2
  for (int s = 0; s < KSTEP / 32; ++s) {
    const size_t k32 = (size_t)s * 32;
    v16h av[MT], bv[NT];
    #pragma unroll
    for (int mi = 0; mi < MT; ++mi) {
      v8h lo = *(const v8h*)(A + aoff[mi] + k32);
      v8h hi = *(const v8h*)(A + aoff[mi] + k32 + 16);
      #pragma unroll
      for (int i = 0; i < 8; ++i) { av[mi][i] = lo[i]; av[mi][i + 8] = hi[i]; }
      __builtin_prefetch(A + aoff[mi] + k32 + 8 * 32, 0, 1);  // global_prefetch_b8
    }
    #pragma unroll
    for (int ni = 0; ni < NT; ++ni) {
      v8h lo = *(const v8h*)(Wh + boff[ni] + k32);
      v8h hi = *(const v8h*)(Wh + boff[ni] + k32 + 16);
      #pragma unroll
      for (int i = 0; i < 8; ++i) { bv[ni][i] = lo[i]; bv[ni][i + 8] = hi[i]; }
      __builtin_prefetch(Wh + boff[ni] + k32 + 8 * 32, 0, 1);
    }
    #pragma unroll
    for (int mi = 0; mi < MT; ++mi)
      #pragma unroll
      for (int ni = 0; ni < NT; ++ni)
        acc[mi][ni] = __builtin_amdgcn_wmma_f32_16x16x32_f16(
            /*neg_a=*/false, av[mi], /*neg_b=*/false, bv[ni],
            /*c_mod=*/(short)0, acc[mi][ni],
            /*reuse_a=*/false, /*reuse_b=*/false);
  }

  // D layout: VGPR d -> row m0 + d + h*8, col n0 + l.
  #pragma unroll
  for (int mi = 0; mi < MT; ++mi) {
    #pragma unroll
    for (int ni = 0; ni < NT; ++ni) {
      float* pp = part + ((size_t)ch * MDIM + (mt0 + mi * 16)) * OSZ
                       + (nt0 + ni * 16);
      #pragma unroll
      for (int d = 0; d < 8; ++d)
        pp[(size_t)(d + h * 8) * OSZ + l] = acc[mi][ni][d];
    }
  }
}

// ---------------------------------------------------------------------------
// 4) Reduce K-chunk partials + bias -> d_out (f32, [MDIM][OSZ]).
// ---------------------------------------------------------------------------
__global__ __launch_bounds__(256) void reduce_kernel(const float* __restrict__ part,
                                                     const float* __restrict__ bias,
                                                     float* __restrict__ out) {
  const int idx = blockIdx.x * 256 + threadIdx.x;     // m*OSZ + n
  float s = bias[idx & (OSZ - 1)];
  #pragma unroll 8
  for (int c = 0; c < KCH; ++c)
    s += part[(size_t)c * (MDIM * OSZ) + idx];
  out[idx] = s;
}

// ---------------------------------------------------------------------------
// Launch.  Inputs (setup_inputs order): x, rxd_perm, e_idx, signs, W, b.
// ws: A f16 @0 (64 MiB) | Wh f16 @64 MiB (64 MiB) | partials f32 @128 MiB (16 MiB)
// ---------------------------------------------------------------------------
extern "C" void kernel_launch(void* const* d_in, const int* in_sizes, int n_in,
                              void* d_out, int out_size, void* d_ws, size_t ws_size,
                              hipStream_t stream) {
  const float* x        = (const float*)d_in[0];
  const int*   rxd_perm = (const int*)  d_in[1];
  const int*   e_idx    = (const int*)  d_in[2];
  const float* signs    = (const float*)d_in[3];
  const float* W        = (const float*)d_in[4];
  const float* bias     = (const float*)d_in[5];
  float*       out      = (float*)d_out;

  char*     ws   = (char*)d_ws;
  _Float16* A    = (_Float16*)ws;                                   // [MDIM][KTOT]
  _Float16* Wh   = (_Float16*)(ws + (size_t)MDIM * KTOT * 2);       // [OSZ ][KTOT]
  float*    part = (float*)   (ws + (size_t)MDIM * KTOT * 4);       // [KCH][MDIM][OSZ]

  const int n4 = (OSZ * KTOT) / 4;                                  // 8388608
  wcvt_kernel<<<(n4 + 255) / 256, 256, 0, stream>>>(W, Wh, n4);

  feat_kernel<<<MDIM * RDIM, 512, 0, stream>>>(x, rxd_perm, e_idx, signs, A);

  gemm_kernel<<<dim3(OSZ / (NT * 16), MDIM / (MT * 16), KCH), 32, 0, stream>>>(A, Wh, part);

  reduce_kernel<<<(MDIM * OSZ) / 256, 256, 0, stream>>>(part, bias, out);
}
